// BleedingDetector_53214644798166
// MI455X (gfx1250) — matrix operand order repaired and checked by reading, not compiled, not run on hardware
//
#include <hip/hip_runtime.h>
#include <hip/hip_bf16.h>

typedef __attribute__((ext_vector_type(16))) _Float16 v16h;
typedef __attribute__((ext_vector_type(8)))  _Float16 v8h;
typedef __attribute__((ext_vector_type(8)))  float    v8f;

#define BATCH 16
#define DT    4
#define HH    21
#define WWID  32
#define HWSZ  672     // 21*32
#define DHW   2688    // 4*672
#define MROWS 43008   // 16*2688
#define EPS_BN 1e-5f

// ---------------------------------------------------------------------------
// features [B,C=512,D,H,W] f32  ->  [B*D*H*W][512] f16 (channels-last)
// ---------------------------------------------------------------------------
__global__ void k_feat2h(const float* __restrict__ feat, _Float16* __restrict__ out) {
  size_t i = (size_t)blockIdx.x * blockDim.x + threadIdx.x;
  const size_t N = (size_t)MROWS * 512;
  if (i >= N) return;
  int c = (int)(i % 512);
  size_t t = i / 512;
  int s = (int)(t % DHW);
  int b = (int)(t / DHW);
  out[i] = (_Float16)feat[((size_t)b * 512 + c) * DHW + s];
}

// ---------------------------------------------------------------------------
// conv weights [O][I][3][3][3] f32 -> [tap][O][I] f16 (K contiguous per lane)
// ---------------------------------------------------------------------------
__global__ void k_wprep(const float* __restrict__ w, _Float16* __restrict__ out,
                        int O, int I) {
  int i = blockIdx.x * blockDim.x + threadIdx.x;
  int total = 27 * O * I;
  if (i >= total) return;
  int ii = i % I;
  int t = i / I;
  int o = t % O;
  int tap = t / O;
  out[i] = (_Float16)w[((size_t)o * I + ii) * 27 + tap];
}

// ---------------------------------------------------------------------------
// helpers: register-only fragment assembly + epilogue store
// ---------------------------------------------------------------------------
__device__ __forceinline__ v16h load_a_frag(const _Float16* p, int k0, bool valid,
                                            v8h z8) {
  // A 16x32 f16 layout: lane holds K {k0+16*lk .. +16} contiguous
  v8h lo = valid ? *(const v8h*)(p + k0)      : z8;
  v8h hi = valid ? *(const v8h*)(p + k0 + 8) : z8;
  return __builtin_shufflevector(lo, hi, 0, 1, 2, 3, 4, 5, 6, 7,
                                 8, 9, 10, 11, 12, 13, 14, 15);
}

template <int COUT>
__device__ __forceinline__ void store_tile(v8f acc, int rbase, int oc,
                                           float inv, float beta, float cb0,
                                           _Float16* __restrict__ Y) {
#pragma unroll
  for (int i = 0; i < 8; ++i) {
    float val = (acc[i] + cb0) * inv + beta;
    val = fmaxf(val, 0.f);
    Y[(long)(rbase + i) * COUT + oc] = (_Float16)val;
  }
}

// ---------------------------------------------------------------------------
// Implicit-GEMM 3x3x3 conv (pad=1) + bias + BN + ReLU via WMMA f16->f32.
//   X  : [MROWS][CIN]  channels-last f16 activations
//   Wt : [27][COUT][CIN] f16 weights
//   Y  : [MROWS][COUT] channels-last f16 output
// Block: 256 threads = 8 waves, tile 64(M) x 128(N); wave tile 32x32.
// Fully SSA hot loop: no arrays/unions so acc/frags stay in VGPRs.
// ---------------------------------------------------------------------------
template <int CIN, int COUT>
__global__ void __launch_bounds__(256) conv_wmma_kernel(
    const _Float16* __restrict__ X, const _Float16* __restrict__ Wt,
    const float* __restrict__ cbias, const float* __restrict__ bg,
    const float* __restrict__ bb, const float* __restrict__ bm,
    const float* __restrict__ bv, _Float16* __restrict__ Y) {
  const int wid  = threadIdx.x >> 5;
  const int lane = threadIdx.x & 31;
  const int lm   = lane & 15;   // A row / B col / C col within 16
  const int lk   = lane >> 4;   // K-half selector
  const int mw   = wid & 1;
  const int nw   = wid >> 1;
  const int Mwave = blockIdx.x * 64 + mw * 32;
  const int Nwave = blockIdx.y * 128 + nw * 32;

  // decompose the two 16-row M subtiles into (b,d,h,w)
  const int r0 = Mwave + lm;
  const int b0 = r0 / DHW; int t0 = r0 - b0 * DHW;
  const int d0 = t0 / HWSZ; t0 -= d0 * HWSZ;
  const int h0 = t0 / WWID; const int w0 = t0 - h0 * WWID;

  const int r1 = Mwave + 16 + lm;
  const int b1 = r1 / DHW; int t1 = r1 - b1 * DHW;
  const int d1 = t1 / HWSZ; t1 -= d1 * HWSZ;
  const int h1 = t1 / WWID; const int w1 = t1 - h1 * WWID;

  v8f acc00 = {}, acc01 = {}, acc10 = {}, acc11 = {};
  const v8h z8 = {};

  for (int tap = 0; tap < 27; ++tap) {
    const int tdd = tap / 9 - 1;
    const int tdh = (tap / 3) % 3 - 1;
    const int tdw = tap % 3 - 1;

    const int di0 = d0 + tdd, hi0 = h0 + tdh, wi0 = w0 + tdw;
    const bool av0 = ((unsigned)di0 < (unsigned)DT) &&
                     ((unsigned)hi0 < (unsigned)HH) &&
                     ((unsigned)wi0 < (unsigned)WWID);
    const _Float16* ap0 =
        X + (((long)b0 * DT + di0) * HWSZ + hi0 * WWID + wi0) * (long)CIN + lk * 16;

    const int di1 = d1 + tdd, hi1 = h1 + tdh, wi1 = w1 + tdw;
    const bool av1 = ((unsigned)di1 < (unsigned)DT) &&
                     ((unsigned)hi1 < (unsigned)HH) &&
                     ((unsigned)wi1 < (unsigned)WWID);
    const _Float16* ap1 =
        X + (((long)b1 * DT + di1) * HWSZ + hi1 * WWID + wi1) * (long)CIN + lk * 16;

    const _Float16* bp0 =
        Wt + ((long)tap * COUT + (Nwave + lm)) * (long)CIN + lk * 16;
    const _Float16* bp1 =
        Wt + ((long)tap * COUT + (Nwave + 16 + lm)) * (long)CIN + lk * 16;

    // prefetch next tap's weight rows into cache (global_prefetch_b8)
    if (tap + 1 < 27) {
      __builtin_prefetch(bp0 + (long)COUT * CIN, 0, 3);
      __builtin_prefetch(bp1 + (long)COUT * CIN, 0, 3);
    }

#pragma unroll 4
    for (int kc = 0; kc < CIN / 32; ++kc) {
      const int k0 = kc * 32;
      v16h a0 = load_a_frag(ap0, k0, av0, z8);
      v16h a1 = load_a_frag(ap1, k0, av1, z8);
      // B 32x16 f16 layout: lane holds col N=lm, contiguous K run of 16 (32B)
      v16h bf0 = *(const v16h*)(bp0 + k0);
      v16h bf1 = *(const v16h*)(bp1 + k0);
      acc00 = __builtin_amdgcn_wmma_f32_16x16x32_f16(false, a0, false, bf0,
                                                     (short)0, acc00, false, false);
      acc01 = __builtin_amdgcn_wmma_f32_16x16x32_f16(false, a0, false, bf1,
                                                     (short)0, acc01, false, false);
      acc10 = __builtin_amdgcn_wmma_f32_16x16x32_f16(false, a1, false, bf0,
                                                     (short)0, acc10, false, false);
      acc11 = __builtin_amdgcn_wmma_f32_16x16x32_f16(false, a1, false, bf1,
                                                     (short)0, acc11, false, false);
    }
  }

  // Fused epilogue: +conv bias, BN affine, ReLU, f16 channels-last store.
  const int oc0 = Nwave + lm;
  const int oc1 = Nwave + 16 + lm;
  const float inv0  = bg[oc0] * rsqrtf(bv[oc0] + EPS_BN);
  const float beta0 = bb[oc0] - bm[oc0] * inv0;
  const float cbv0  = cbias[oc0];
  const float inv1  = bg[oc1] * rsqrtf(bv[oc1] + EPS_BN);
  const float beta1 = bb[oc1] - bm[oc1] * inv1;
  const float cbv1  = cbias[oc1];
  const int rbase0 = Mwave + 8 * lk;        // mi = 0 rows
  const int rbase1 = Mwave + 16 + 8 * lk;   // mi = 1 rows
  store_tile<COUT>(acc00, rbase0, oc0, inv0, beta0, cbv0, Y);
  store_tile<COUT>(acc01, rbase0, oc1, inv1, beta1, cbv1, Y);
  store_tile<COUT>(acc10, rbase1, oc0, inv0, beta0, cbv0, Y);
  store_tile<COUT>(acc11, rbase1, oc1, inv1, beta1, cbv1, Y);
}

// ---------------------------------------------------------------------------
// 1x1x1 conv (128->8) + mean over time -> raw [B][672][8] f32
// ---------------------------------------------------------------------------
__global__ void k_conv3(const _Float16* __restrict__ X2,
                        const float* __restrict__ w3,
                        const float* __restrict__ b3,
                        float* __restrict__ raw) {
  int idx = blockIdx.x * blockDim.x + threadIdx.x;   // [0, 16*672*8)
  if (idx >= BATCH * HWSZ * 8) return;
  int co = idx & 7;
  int t = idx >> 3;
  int n = t % HWSZ;
  int b = t / HWSZ;
  float acc = 0.f;
  const float* wp = w3 + co * 128;
  for (int d = 0; d < DT; ++d) {
    const _Float16* p = X2 + (((size_t)(b * DT + d) * HWSZ) + n) * 128;
    for (int ci = 0; ci < 128; ++ci) acc += (float)p[ci] * wp[ci];
  }
  raw[idx] = 0.25f * acc + b3[co];
}

// ---------------------------------------------------------------------------
// Global average pool over D*H*W: one wave per (b,c)
// ---------------------------------------------------------------------------
__global__ void k_pool(const float* __restrict__ feat, float* __restrict__ pooled) {
  int wid = (int)((blockIdx.x * blockDim.x + threadIdx.x) >> 5);
  int lane = threadIdx.x & 31;
  if (wid >= BATCH * 512) return;
  const float* p = feat + (size_t)wid * DHW;
  float s = 0.f;
  for (int j = lane; j < DHW; j += 32) s += p[j];
#pragma unroll
  for (int o = 16; o > 0; o >>= 1) s += __shfl_down(s, o, 32);
  if (lane == 0) pooled[wid] = s * (1.f / (float)DHW);
}

// ---------------------------------------------------------------------------
// clip (2) + severity (4) linear heads; one block per batch, 512 threads
// out layout: clip [0,32) | sev [32,96)
// ---------------------------------------------------------------------------
__global__ void k_heads(const float* __restrict__ pooled,
                        const float* __restrict__ cw, const float* __restrict__ cb,
                        const float* __restrict__ sw, const float* __restrict__ sb,
                        float* __restrict__ out) {
  int b = blockIdx.x;
  int tid = threadIdx.x;               // 512 threads
  __shared__ float red[512];
  float p = pooled[b * 512 + tid];
  for (int j = 0; j < 6; ++j) {
    const float* wr = (j < 2) ? (cw + j * 512) : (sw + (j - 2) * 512);
    red[tid] = p * wr[tid];
    __syncthreads();
    for (int s = 256; s > 0; s >>= 1) {
      if (tid < s) red[tid] += red[tid + s];
      __syncthreads();
    }
    if (tid == 0) {
      if (j < 2) out[b * 2 + j] = red[0] + cb[j];
      else       out[32 + b * 4 + (j - 2)] = red[0] + sb[j - 2];
    }
    __syncthreads();
  }
}

// ---------------------------------------------------------------------------
// YOLO-style decode: raw [B*672][8] -> boxes/conf/class
// ---------------------------------------------------------------------------
__global__ void k_decode(const float* __restrict__ raw,
                         float* __restrict__ boxes, float* __restrict__ conf,
                         int* __restrict__ cls) {
  int idx = blockIdx.x * blockDim.x + threadIdx.x;   // [0, 16*672)
  if (idx >= BATCH * HWSZ) return;
  int n = idx % HWSZ;
  int h = n / WWID, w = n % WWID;
  const float* r = raw + (size_t)idx * 8;
  float xo = 1.f / (1.f + expf(-r[0]));
  float yo = 1.f / (1.f + expf(-r[1]));
  float ws = expf(r[2]);
  float hs = expf(r[3]);
  float obj = 1.f / (1.f + expf(-r[4]));
  float mx = fmaxf(r[5], fmaxf(r[6], r[7]));
  float e0 = expf(r[5] - mx), e1 = expf(r[6] - mx), e2 = expf(r[7] - mx);
  float inv = 1.f / (e0 + e1 + e2);
  float p0 = e0 * inv, p1 = e1 * inv, p2 = e2 * inv;
  int bi = 0; float bp = p0;
  if (p1 > bp) { bp = p1; bi = 1; }
  if (p2 > bp) { bp = p2; bi = 2; }
  float gx = (float)w / (float)WWID;
  float gy = (float)h / (float)HH;
  float xc = (gx + xo) * 512.f;   // IN_W
  float yc = (gy + yo) * 328.f;   // IN_H
  float bw = 0.2f * ws * 512.f;   // ANCHOR_W * IN_W
  float bh = 0.29f * hs * 328.f;  // ANCHOR_H * IN_H
  boxes[(size_t)idx * 4 + 0] = xc - bw * 0.5f;
  boxes[(size_t)idx * 4 + 1] = yc - bh * 0.5f;
  boxes[(size_t)idx * 4 + 2] = xc + bw * 0.5f;
  boxes[(size_t)idx * 4 + 3] = yc + bh * 0.5f;
  conf[idx] = obj * bp;
  cls[idx]  = bi;
}

// ---------------------------------------------------------------------------
// Greedy NMS, K=10, one block per batch. Matches jnp.argmax first-max ties.
// out layout: boxes [96,736) | scores [736,896) | classes [896,1056)
// ---------------------------------------------------------------------------
__global__ void k_nms(const float* __restrict__ boxes, const float* __restrict__ conf,
                      const int* __restrict__ cls, float* __restrict__ out) {
  int b = blockIdx.x;
  int tid = threadIdx.x;   // 256 threads
  __shared__ float sc[HWSZ];
  __shared__ float rv[256];
  __shared__ int   ri[256];
  __shared__ float sbox[4];
  __shared__ float ss;
  __shared__ int   sidx;
  const float* bx = boxes + (size_t)b * HWSZ * 4;
  for (int j = tid; j < HWSZ; j += 256) {
    float c = conf[b * HWSZ + j];
    sc[j] = (c > 0.2f) ? c : -1.f;
  }
  __syncthreads();
  float* oBoxes  = out + 96 + b * 40;
  float* oScores = out + 736 + b * 10;
  float* oCls    = out + 896 + b * 10;
  for (int it = 0; it < 10; ++it) {
    float bv = -1e30f; int bi = 1 << 30;
    for (int j = tid; j < HWSZ; j += 256) {
      float v = sc[j];
      if (v > bv || (v == bv && j < bi)) { bv = v; bi = j; }
    }
    rv[tid] = bv; ri[tid] = bi;
    __syncthreads();
    for (int s = 128; s > 0; s >>= 1) {
      if (tid < s) {
        if (rv[tid + s] > rv[tid] ||
            (rv[tid + s] == rv[tid] && ri[tid + s] < ri[tid])) {
          rv[tid] = rv[tid + s]; ri[tid] = ri[tid + s];
        }
      }
      __syncthreads();
    }
    if (tid == 0) {
      sidx = ri[0]; ss = rv[0];
      for (int q = 0; q < 4; ++q) sbox[q] = bx[ri[0] * 4 + q];
    }
    __syncthreads();
    int i = sidx; float s_ = ss;
    if (tid == 0) {
      float vf = (s_ > 0.f) ? 1.f : 0.f;
      for (int q = 0; q < 4; ++q) oBoxes[it * 4 + q] = sbox[q] * vf;
      oScores[it] = fmaxf(s_, 0.f) * vf;
      oCls[it] = (s_ > 0.f) ? (float)cls[b * HWSZ + i] : -1.f;
    }
    float x1 = sbox[0], y1 = sbox[1], x2 = sbox[2], y2 = sbox[3];
    float a1 = (x2 - x1) * (y2 - y1);
    for (int j = tid; j < HWSZ; j += 256) {
      float jx1 = bx[j * 4 + 0], jy1 = bx[j * 4 + 1];
      float jx2 = bx[j * 4 + 2], jy2 = bx[j * 4 + 3];
      float ix1 = fmaxf(x1, jx1), iy1 = fmaxf(y1, jy1);
      float ix2 = fminf(x2, jx2), iy2 = fminf(y2, jy2);
      float inter = fmaxf(ix2 - ix1, 0.f) * fmaxf(iy2 - iy1, 0.f);
      float a2 = (jx2 - jx1) * (jy2 - jy1);
      float iou = inter / fmaxf(a1 + a2 - inter, 1e-9f);
      if (iou > 0.3f || j == i) sc[j] = -1.f;
    }
    __syncthreads();
  }
}

// ---------------------------------------------------------------------------
// Workspace layout (bytes)
// ---------------------------------------------------------------------------
static constexpr size_t OFF_FEAT = 0;                                     // f16 [M][512]
static constexpr size_t OFF_W1   = OFF_FEAT + (size_t)MROWS * 512 * 2;    // f16 [27][256][512]
static constexpr size_t OFF_X1   = OFF_W1 + (size_t)27 * 256 * 512 * 2;   // f16 [M][256]
static constexpr size_t OFF_W2   = OFF_X1 + (size_t)MROWS * 256 * 2;      // f16 [27][128][256]
static constexpr size_t OFF_X2   = OFF_W2 + (size_t)27 * 128 * 256 * 2;   // f16 [M][128]
static constexpr size_t OFF_RAW  = OFF_X2 + (size_t)MROWS * 128 * 2;      // f32 [16][672][8]
static constexpr size_t OFF_POOL = OFF_RAW + (size_t)BATCH * HWSZ * 8 * 4;// f32 [16][512]
static constexpr size_t OFF_BOX  = OFF_POOL + (size_t)BATCH * 512 * 4;    // f32 [16][672][4]
static constexpr size_t OFF_CONF = OFF_BOX + (size_t)BATCH * HWSZ * 4 * 4;// f32 [16][672]
static constexpr size_t OFF_CLS  = OFF_CONF + (size_t)BATCH * HWSZ * 4;   // i32 [16][672]

extern "C" void kernel_launch(void* const* d_in, const int* in_sizes, int n_in,
                              void* d_out, int out_size, void* d_ws, size_t ws_size,
                              hipStream_t stream) {
  const float* features = (const float*)d_in[0];
  const float* conv1_w  = (const float*)d_in[1];
  const float* conv1_b  = (const float*)d_in[2];
  const float* bn1_g    = (const float*)d_in[3];
  const float* bn1_b    = (const float*)d_in[4];
  const float* bn1_m    = (const float*)d_in[5];
  const float* bn1_v    = (const float*)d_in[6];
  const float* conv2_w  = (const float*)d_in[7];
  const float* conv2_b  = (const float*)d_in[8];
  const float* bn2_g    = (const float*)d_in[9];
  const float* bn2_b    = (const float*)d_in[10];
  const float* bn2_m    = (const float*)d_in[11];
  const float* bn2_v    = (const float*)d_in[12];
  const float* conv3_w  = (const float*)d_in[13];
  const float* conv3_b  = (const float*)d_in[14];
  const float* cls_w    = (const float*)d_in[15];
  const float* cls_b    = (const float*)d_in[16];
  const float* sev_w    = (const float*)d_in[17];
  const float* sev_b    = (const float*)d_in[18];

  char* ws = (char*)d_ws;
  _Float16* featH = (_Float16*)(ws + OFF_FEAT);
  _Float16* w1t   = (_Float16*)(ws + OFF_W1);
  _Float16* x1    = (_Float16*)(ws + OFF_X1);
  _Float16* w2t   = (_Float16*)(ws + OFF_W2);
  _Float16* x2    = (_Float16*)(ws + OFF_X2);
  float*    raw   = (float*)(ws + OFF_RAW);
  float*    pool  = (float*)(ws + OFF_POOL);
  float*    dbox  = (float*)(ws + OFF_BOX);
  float*    dconf = (float*)(ws + OFF_CONF);
  int*      dcls  = (int*)(ws + OFF_CLS);
  float*    out   = (float*)d_out;

  // 1. data layout transforms (f32 -> channels-last / O-major f16)
  {
    size_t n = (size_t)MROWS * 512;
    k_feat2h<<<dim3((unsigned)((n + 255) / 256)), dim3(256), 0, stream>>>(features, featH);
  }
  k_wprep<<<dim3((27 * 256 * 512 + 255) / 256), dim3(256), 0, stream>>>(conv1_w, w1t, 256, 512);
  k_wprep<<<dim3((27 * 128 * 256 + 255) / 256), dim3(256), 0, stream>>>(conv2_w, w2t, 128, 256);

  // 2. WMMA implicit-GEMM convs with fused bias+BN+ReLU
  conv_wmma_kernel<512, 256><<<dim3(MROWS / 64, 2), dim3(256), 0, stream>>>(
      featH, w1t, conv1_b, bn1_g, bn1_b, bn1_m, bn1_v, x1);
  conv_wmma_kernel<256, 128><<<dim3(MROWS / 64, 1), dim3(256), 0, stream>>>(
      x1, w2t, conv2_b, bn2_g, bn2_b, bn2_m, bn2_v, x2);

  // 3. 1x1x1 conv + time-mean
  k_conv3<<<dim3((BATCH * HWSZ * 8 + 255) / 256), dim3(256), 0, stream>>>(
      x2, conv3_w, conv3_b, raw);

  // 4. classification heads
  k_pool<<<dim3((BATCH * 512 * 32 + 255) / 256), dim3(256), 0, stream>>>(features, pool);
  k_heads<<<dim3(BATCH), dim3(512), 0, stream>>>(pool, cls_w, cls_b, sev_w, sev_b, out);

  // 5. decode + NMS
  k_decode<<<dim3((BATCH * HWSZ + 255) / 256), dim3(256), 0, stream>>>(raw, dbox, dconf, dcls);
  k_nms<<<dim3(BATCH), dim3(256), 0, stream>>>(dbox, dconf, dcls, out);

  (void)in_sizes; (void)n_in; (void)out_size; (void)ws_size;
}
